// Qwen3MoeDecoderLayer_49091476193832
// MI455X (gfx1250) — compile-verified, matrix operand order checked by reading, not run on hardware
//
#include <hip/hip_runtime.h>
#include <hip/hip_bf16.h>
#include <cstdint>

// ---------------------------------------------------------------------------
// Qwen3-MoE decoder layer for MI455X (gfx1250, wave32, WMMA).
// All matmuls use v_wmma_f32_16x16x32_bf16 (16x16 tiles, K=32).
// Straight-copy LDS stagings use GLOBAL_LOAD_ASYNC_TO_LDS_B128 (ASYNCcnt).
// ---------------------------------------------------------------------------

typedef __bf16 bf16;
typedef __attribute__((ext_vector_type(16))) __bf16 v16bf;
typedef __attribute__((ext_vector_type(8)))  __bf16 v8bf;
typedef __attribute__((ext_vector_type(8)))  float   v8f;

#define CB  2
#define CS  2048
#define CH  2048
#define CNH 16
#define CNKV 4
#define CD  128
#define CE  16
#define CK  2
#define CF  768
#define CT  (CB * CS)     // 4096 tokens
#define CP  (CT * CK)     // 8192 assignments
#define CC  1024          // capacity = 2*P/E

union FragBF { v16bf v; v8bf h[2]; };

__device__ __forceinline__ v8f v8f_zero() {
    v8f z;
#pragma unroll
    for (int i = 0; i < 8; ++i) z[i] = 0.f;
    return z;
}

// 32-bit LDS byte offset of a shared-memory object (for async VDST operand).
__device__ __forceinline__ uint32_t lds_off(const void* p) {
    return (uint32_t)(uintptr_t)(__attribute__((address_space(3))) const void*)p;
}

// Async global -> LDS copy, 16 bytes per lane.  Tracked with ASYNCcnt.
__device__ __forceinline__ void async_copy_b128(uint32_t dst_lds, const void* src) {
    asm volatile("global_load_async_to_lds_b128 %0, %1, off"
                 :: "v"(dst_lds), "v"(src) : "memory");
}
__device__ __forceinline__ void wait_async() {
    asm volatile("s_wait_asynccnt 0x0" ::: "memory");
}

// Load one 16x32 bf16 A/B fragment from a row-major LDS tile.
// Layout (ISA 7.12.2, 16-bit A 16x32): lanes 0-15 hold K 0..7 & 16..23,
// lanes 16-31 hold K 8..15 & 24..31, for row = lane&15 (B: n = lane&15 from
// a [n][k]-transposed tile).
__device__ __forceinline__ v16bf frag_ld(const bf16* base, int ldm, int row,
                                         int kOff, int lane) {
    const int hi = (lane >> 4) & 1;
    const bf16* p = base + (size_t)row * ldm + kOff + hi * 8;
    FragBF f;
    f.h[0] = *(const v8bf*)(p);
    f.h[1] = *(const v8bf*)(p + 16);
    return f.v;
}

// ---------------------------------------------------------------------------
// Fused (a + b) -> resid ; rmsnorm(resid)*w -> bf16 h.   One block per token.
// ---------------------------------------------------------------------------
__global__ __launch_bounds__(256)
void add_rmsnorm_kernel(const float* __restrict__ a, const float* __restrict__ b,
                        const float* __restrict__ w, float* __restrict__ resid,
                        bf16* __restrict__ hout) {
    const int t = blockIdx.x, tid = threadIdx.x;
    __shared__ float red[256];
    const long long base = (long long)t * CH;
    float x[8]; float ss = 0.f;
#pragma unroll
    for (int j = 0; j < 8; ++j) {
        int c = tid * 8 + j;
        float v = a[base + c] + b[base + c];
        x[j] = v; ss += v * v;
    }
    red[tid] = ss; __syncthreads();
    for (int s = 128; s > 0; s >>= 1) {
        if (tid < s) red[tid] += red[tid + s];
        __syncthreads();
    }
    const float inv = rsqrtf(red[0] / (float)CH + 1e-6f);
#pragma unroll
    for (int j = 0; j < 8; ++j) {
        int c = tid * 8 + j;
        resid[base + c] = x[j];
        hout[base + c] = (bf16)(x[j] * inv * w[c]);
    }
}

// ---------------------------------------------------------------------------
__global__ void f32_to_bf16_kernel(const float* __restrict__ src,
                                   bf16* __restrict__ dst, long long n) {
    long long i = (long long)blockIdx.x * blockDim.x + threadIdx.x;
    const long long stride = (long long)gridDim.x * blockDim.x;
    for (; i < n; i += stride) dst[i] = (bf16)src[i];
}

// ---------------------------------------------------------------------------
// Generic batched bf16 WMMA GEMM:  C[M,N](f32) = A[M,K] * B[K,N]  (row major).
// 64x64 tile per 256-thread block (8 waves, each wave -> one 16x32 C piece).
// A tile staged via async global->LDS; B tile transposed in flight.
// ---------------------------------------------------------------------------
#define GLDS 40   // 32 K + 8 pad (80B rows -> 16B aligned frag reads)

__global__ __launch_bounds__(256)
void gemm_bf16_wmma(const bf16* __restrict__ A, const bf16* __restrict__ Bw,
                    float* __restrict__ C, int M, int N, int Kd,
                    long long sAe, long long sBe, long long sCe) {
    __shared__ bf16 sA[64][GLDS];
    __shared__ bf16 sB[64][GLDS];   // stored transposed: [n][k]
    const int tid = threadIdx.x, lane = tid & 31, w = tid >> 5;
    const int lo = lane & 15, hi = lane >> 4;
    A  += (long long)blockIdx.z * sAe;
    Bw += (long long)blockIdx.z * sBe;
    C  += (long long)blockIdx.z * sCe;
    const int m0 = blockIdx.y * 64, n0 = blockIdx.x * 64;
    const int wm = (w & 3) * 16, wn = (w >> 2) * 32;

    v8f acc0 = v8f_zero(), acc1 = v8f_zero();
    const int arow = tid >> 2, acol = (tid & 3) * 8;   // A tile 64x32
    const int brow = tid >> 3, bcol = (tid & 7) * 8;   // B tile 32x64
    const uint32_t aLds = lds_off(&sA[arow][acol]);

    for (int k0 = 0; k0 < Kd; k0 += 32) {
        if (k0 + 32 < Kd) {   // prefetch next K tile -> global_prefetch_b8
            __builtin_prefetch(A  + (long long)(m0 + arow) * Kd + k0 + 32 + acol, 0, 1);
            __builtin_prefetch(Bw + (long long)(k0 + 32 + brow) * N + n0 + bcol, 0, 1);
        }
        // A tile: async DMA straight into LDS (no VGPR round trip)
        async_copy_b128(aLds, A + (long long)(m0 + arow) * Kd + k0 + acol);
        // B tile: load + transpose into LDS
        v8bf bv = *(const v8bf*)(Bw + (long long)(k0 + brow) * N + n0 + bcol);
#pragma unroll
        for (int j = 0; j < 8; ++j) sB[bcol + j][brow] = bv[j];
        wait_async();
        __syncthreads();
        v16bf af  = frag_ld(&sA[0][0], GLDS, wm + lo,      0, lane);
        v16bf bf0 = frag_ld(&sB[0][0], GLDS, wn + lo,      0, lane);
        v16bf bf1 = frag_ld(&sB[0][0], GLDS, wn + 16 + lo, 0, lane);
        acc0 = __builtin_amdgcn_wmma_f32_16x16x32_bf16(false, af, false, bf0,
                                                       (short)0, acc0, false, false);
        acc1 = __builtin_amdgcn_wmma_f32_16x16x32_bf16(false, af, false, bf1,
                                                       (short)0, acc1, false, false);
        __syncthreads();
    }
#pragma unroll
    for (int r = 0; r < 8; ++r) {
        const long long row = m0 + wm + r + hi * 8;
        C[row * N + n0 + wn + lo]      = acc0[r];
        C[row * N + n0 + wn + 16 + lo] = acc1[r];
    }
}

// ---------------------------------------------------------------------------
// Per-head RMSNorm + RoPE, fp32 [T][nh*128] -> bf16 [B][nh][S][128].
// ---------------------------------------------------------------------------
__global__ __launch_bounds__(128)
void qk_norm_rope_kernel(const float* __restrict__ src, const float* __restrict__ nw,
                         const int* __restrict__ positions, bf16* __restrict__ dst,
                         int nh) {
    const int gb = blockIdx.x;
    const int t = gb / nh, head = gb % nh, d = threadIdx.x;
    const int b = t >> 11, s = t & (CS - 1);
    __shared__ float red[128];
    __shared__ float xn[128];
    float x = src[(long long)t * (nh * CD) + head * CD + d];
    red[d] = x * x; __syncthreads();
    for (int st = 64; st > 0; st >>= 1) {
        if (d < st) red[d] += red[d + st];
        __syncthreads();
    }
    const float inv = rsqrtf(red[0] / (float)CD + 1e-6f);
    xn[d] = x * inv * nw[d];
    __syncthreads();
    const long long ob = (((long long)(b * nh + head)) * CS + s) * CD;
    if (d < 64) {
        float fpos = (float)positions[s];
        float invf = __expf(-((float)d / 64.0f) * 13.815510557964274f); // ln(1e6)
        float ang = fpos * invf;
        float c = __cosf(ang), sn = __sinf(ang);
        float x1 = xn[d], x2 = xn[d + 64];
        dst[ob + d]      = (bf16)(x1 * c - x2 * sn);
        dst[ob + d + 64] = (bf16)(x2 * c + x1 * sn);
    }
}

__global__ __launch_bounds__(128)
void v_reorder_kernel(const float* __restrict__ src, bf16* __restrict__ dst) {
    const int gb = blockIdx.x;                 // T * NKV
    const int t = gb >> 2, head = gb & 3, d = threadIdx.x;
    const int b = t >> 11, s = t & (CS - 1);
    dst[(((long long)(b * CNKV + head)) * CS + s) * CD + d] =
        (bf16)src[(long long)t * (CNKV * CD) + head * CD + d];
}

// ---------------------------------------------------------------------------
// Flash attention, causal, GQA (4 q-heads per kv-head).
// 128 q rows / block (8 waves x 16 rows), 64 kv rows per inner tile.
// QK^T and P*V both via v_wmma_f32_16x16x32_bf16; online softmax lives in the
// accumulator's natural (vgpr=row, lane-half) layout.  Q/K tiles staged with
// async global->LDS DMA; V transposed in flight.
// ---------------------------------------------------------------------------
#define QSTR 136
#define VSTR 72
#define PSTR 72

__global__ __launch_bounds__(256)
void flash_attn_kernel(const bf16* __restrict__ Q, const bf16* __restrict__ Kg,
                       const bf16* __restrict__ Vg, bf16* __restrict__ Og) {
    __shared__ bf16 sQ[128][QSTR];
    __shared__ bf16 sK[64][QSTR];
    __shared__ bf16 sVt[128][VSTR];        // transposed: [d][kv]
    __shared__ bf16 sP[8][16][PSTR];       // per-wave P staging
    const int tid = threadIdx.x, lane = tid & 31, w = tid >> 5;
    const int lo = lane & 15, hi = lane >> 4;
    const int q0 = blockIdx.x * 128;
    const int b = blockIdx.y / CNH, h = blockIdx.y % CNH, kvh = h >> 2;

    const bf16* Qb = Q  + (((long long)(b * CNH + h)) * CS + q0) * CD;
    const bf16* Kb = Kg + ((long long)(b * CNKV + kvh)) * CS * CD;
    const bf16* Vb = Vg + ((long long)(b * CNKV + kvh)) * CS * CD;

#pragma unroll
    for (int rep = 0; rep < 8; ++rep) {    // Q tile 128x128 via async DMA
        int lin = (rep * 256 + tid) * 8;
        int row = lin >> 7, col = lin & 127;
        async_copy_b128(lds_off(&sQ[row][col]), Qb + (long long)row * CD + col);
    }
    wait_async();
    __syncthreads();
    const int wm = w * 16;
    v16bf qf[4];
#pragma unroll
    for (int kk = 0; kk < 4; ++kk)
        qf[kk] = frag_ld(&sQ[0][0], QSTR, wm + lo, kk * 32, lane);

    float m_i[8], l_i[8];
    v8f oacc[8];
#pragma unroll
    for (int r = 0; r < 8; ++r) { m_i[r] = -1e30f; l_i[r] = 0.f; }
#pragma unroll
    for (int j = 0; j < 8; ++j) oacc[j] = v8f_zero();

    const float scale = 0.0883883476483184f;   // 1/sqrt(128)
    const int ktMax = (q0 + 127) >> 6;

    for (int kt = 0; kt <= ktMax; ++kt) {
#pragma unroll
        for (int rep = 0; rep < 4; ++rep) {    // K,V tiles 64x128
            int lin = (rep * 256 + tid) * 8;
            int row = lin >> 7, col = lin & 127;
            async_copy_b128(lds_off(&sK[row][col]),
                            Kb + ((long long)(kt * 64 + row)) * CD + col);
            v8bf vv = *(const v8bf*)(Vb + ((long long)(kt * 64 + row)) * CD + col);
#pragma unroll
            for (int j = 0; j < 8; ++j) sVt[col + j][row] = vv[j];
        }
        wait_async();
        __syncthreads();

        v8f sacc[4];
#pragma unroll
        for (int ns = 0; ns < 4; ++ns) {
            sacc[ns] = v8f_zero();
#pragma unroll
            for (int kk = 0; kk < 4; ++kk) {
                v16bf kf = frag_ld(&sK[0][0], QSTR, ns * 16 + lo, kk * 32, lane);
                sacc[ns] = __builtin_amdgcn_wmma_f32_16x16x32_bf16(
                    false, qf[kk], false, kf, (short)0, sacc[ns], false, false);
            }
        }
        // scale + causal mask
#pragma unroll
        for (int ns = 0; ns < 4; ++ns) {
            const int col = kt * 64 + ns * 16 + lo;
#pragma unroll
            for (int r = 0; r < 8; ++r) {
                const int qrow = q0 + wm + r + hi * 8;
                float sv = sacc[ns][r] * scale;
                sacc[ns][r] = (col <= qrow) ? sv : -1e30f;
            }
        }
        // online softmax: per-row max / sum via 16-lane half-wave shuffles
        float nm[8], pscale[8];
#pragma unroll
        for (int r = 0; r < 8; ++r) {
            float tm = fmaxf(fmaxf(sacc[0][r], sacc[1][r]),
                             fmaxf(sacc[2][r], sacc[3][r]));
#pragma unroll
            for (int off = 8; off >= 1; off >>= 1)
                tm = fmaxf(tm, __shfl_xor(tm, off, 32));
            nm[r] = fmaxf(m_i[r], tm);
            pscale[r] = __expf(m_i[r] - nm[r]);
        }
#pragma unroll
        for (int r = 0; r < 8; ++r) {
            float rs = 0.f;
#pragma unroll
            for (int ns = 0; ns < 4; ++ns) {
                float pv = __expf(sacc[ns][r] - nm[r]);
                sacc[ns][r] = pv; rs += pv;
            }
#pragma unroll
            for (int off = 8; off >= 1; off >>= 1)
                rs += __shfl_xor(rs, off, 32);
            l_i[r] = l_i[r] * pscale[r] + rs;
            m_i[r] = nm[r];
#pragma unroll
            for (int j = 0; j < 8; ++j) oacc[j][r] *= pscale[r];
#pragma unroll
            for (int ns = 0; ns < 4; ++ns)
                sP[w][r + hi * 8][ns * 16 + lo] = (bf16)sacc[ns][r];
        }
        __syncthreads();
        // O += P * V
#pragma unroll
        for (int kk2 = 0; kk2 < 2; ++kk2) {
            v16bf pf = frag_ld(&sP[w][0][0], PSTR, lo, kk2 * 32, lane);
#pragma unroll
            for (int nd = 0; nd < 8; ++nd) {
                v16bf vf = frag_ld(&sVt[0][0], VSTR, nd * 16 + lo, kk2 * 32, lane);
                oacc[nd] = __builtin_amdgcn_wmma_f32_16x16x32_bf16(
                    false, pf, false, vf, (short)0, oacc[nd], false, false);
            }
        }
        __syncthreads();
    }
#pragma unroll
    for (int r = 0; r < 8; ++r) {
        const float inv = 1.0f / l_i[r];
        const long long t = (long long)b * CS + (q0 + wm + r + hi * 8);
#pragma unroll
        for (int nd = 0; nd < 8; ++nd)
            Og[t * (CNH * CD) + h * CD + nd * 16 + lo] = (bf16)(oacc[nd][r] * inv);
    }
}

// ---------------------------------------------------------------------------
// MoE: gate logits + softmax + top-2 (one block / token).
// ---------------------------------------------------------------------------
__global__ __launch_bounds__(256)
void gate_topk_kernel(const bf16* __restrict__ x, const float* __restrict__ gw,
                      int* __restrict__ tids, float* __restrict__ tw) {
    const int t = blockIdx.x, tid = threadIdx.x;
    const int e = tid & 15, part = tid >> 4;
    __shared__ float acc[256];
    __shared__ float logits[16];
    float s = 0.f;
    for (int i = 0; i < 128; ++i) {
        int c = part * 128 + i;
        s += (float)x[(long long)t * CH + c] * gw[c * CE + e];
    }
    acc[tid] = s; __syncthreads();
    if (tid < 16) {
        float tot = 0.f;
        for (int p = 0; p < 16; ++p) tot += acc[p * 16 + tid];
        logits[tid] = tot;
    }
    __syncthreads();
    if (tid == 0) {
        float mx = logits[0];
        for (int i = 1; i < CE; ++i) mx = fmaxf(mx, logits[i]);
        float pr[CE]; float sum = 0.f;
        for (int i = 0; i < CE; ++i) { pr[i] = __expf(logits[i] - mx); sum += pr[i]; }
        for (int i = 0; i < CE; ++i) pr[i] /= sum;
        int i1 = 0;
        for (int i = 1; i < CE; ++i) if (pr[i] > pr[i1]) i1 = i;
        int i2 = (i1 == 0) ? 1 : 0;
        for (int i = 0; i < CE; ++i) if (i != i1 && pr[i] > pr[i2]) i2 = i;
        const float wsum = pr[i1] + pr[i2];
        tids[t * CK] = i1; tids[t * CK + 1] = i2;
        tw[t * CK] = pr[i1] / wsum; tw[t * CK + 1] = pr[i2] / wsum;
    }
}

// Sequential capacity routing (matches reference cumsum ordering).
__global__ void route_experts_kernel(const int* __restrict__ tids,
                                     int* __restrict__ es, int* __restrict__ pc) {
    if (blockIdx.x == 0 && threadIdx.x == 0) {
        int counts[CE];
        for (int i = 0; i < CE; ++i) counts[i] = 0;
        for (int p = 0; p < CP; ++p) {
            int e = tids[p];
            int pos = counts[e]++;
            pc[p] = (pos < CC) ? pos : (CC - 1);
            es[p] = (pos < CC) ? e : CE;
        }
    }
}

__global__ __launch_bounds__(256)
void gather_tokens_kernel(const bf16* __restrict__ x, const int* __restrict__ es,
                          const int* __restrict__ pc, bf16* __restrict__ buf) {
    const int p = blockIdx.x;
    const int e = es[p];
    if (e >= CE) return;
    const long long dst = ((long long)e * CC + pc[p]) * CH;
    const long long src = ((long long)(p >> 1)) * CH;
    const int col = threadIdx.x * 8;
    *(v8bf*)&buf[dst + col] = *(const v8bf*)&x[src + col];
}

__global__ void silu_mul_kernel(const float* __restrict__ gu, bf16* __restrict__ act) {
    const long long n = (long long)CE * CC * CF;
    long long i = (long long)blockIdx.x * blockDim.x + threadIdx.x;
    if (i >= n) return;
    const long long ec = i / CF;
    const int f = (int)(i % CF);
    const float g = gu[ec * (2 * CF) + f];
    const float u = gu[ec * (2 * CF) + CF + f];
    act[i] = (bf16)((g / (1.0f + __expf(-g))) * u);
}

__global__ __launch_bounds__(256)
void moe_combine_kernel(const float* __restrict__ dn, const int* __restrict__ es,
                        const int* __restrict__ pc, const float* __restrict__ tw,
                        float* __restrict__ y) {
    const int t = blockIdx.x;
    const int col = threadIdx.x * 8;
    float accv[8];
#pragma unroll
    for (int j = 0; j < 8; ++j) accv[j] = 0.f;
    for (int k = 0; k < CK; ++k) {
        const int p = t * CK + k;
        const int e = es[p];
        if (e < CE) {
            const float wgt = tw[p];
            const float* row = dn + ((long long)e * CC + pc[p]) * CH;
#pragma unroll
            for (int j = 0; j < 8; ++j) accv[j] += wgt * row[col + j];
        }
    }
#pragma unroll
    for (int j = 0; j < 8; ++j) y[(long long)t * CH + col + j] = accv[j];
}

// ---------------------------------------------------------------------------
extern "C" void kernel_launch(void* const* d_in, const int* in_sizes, int n_in,
                              void* d_out, int out_size, void* d_ws, size_t ws_size,
                              hipStream_t stream) {
    const int*   positions = (const int*)d_in[0];
    const float* hidden    = (const float*)d_in[1];
    const float* residual  = (const float*)d_in[2];
    const float* w_in_ln   = (const float*)d_in[3];
    const float* wq        = (const float*)d_in[4];
    const float* wk        = (const float*)d_in[5];
    const float* wv        = (const float*)d_in[6];
    const float* wo        = (const float*)d_in[7];
    const float* q_norm_w  = (const float*)d_in[8];
    const float* k_norm_w  = (const float*)d_in[9];
    const float* w_post_ln = (const float*)d_in[10];
    const float* gate_w    = (const float*)d_in[11];
    const float* w_gate_up = (const float*)d_in[12];
    const float* w_down    = (const float*)d_in[13];

    float* y_out   = (float*)d_out;
    float* res_out = y_out + (long long)CT * CH;

    // bump allocator over d_ws
    char* wsp = (char*)d_ws;
    auto alloc = [&](size_t bytes) -> void* {
        void* p = (void*)wsp;
        wsp += (bytes + 255) & ~(size_t)255;
        return p;
    };
    bf16*  h_bf   = (bf16*)alloc((size_t)CT * CH * 2);
    bf16*  wq_bf  = (bf16*)alloc((size_t)CH * (CNH * CD) * 2);
    bf16*  wk_bf  = (bf16*)alloc((size_t)CH * (CNKV * CD) * 2);
    bf16*  wv_bf  = (bf16*)alloc((size_t)CH * (CNKV * CD) * 2);
    bf16*  wo_bf  = (bf16*)alloc((size_t)(CNH * CD) * CH * 2);
    bf16*  wgu_bf = (bf16*)alloc((size_t)CE * CH * (2 * CF) * 2);
    bf16*  wdn_bf = (bf16*)alloc((size_t)CE * CF * CH * 2);
    float* q_raw  = (float*)alloc((size_t)CT * (CNH * CD) * 4);
    float* k_raw  = (float*)alloc((size_t)CT * (CNKV * CD) * 4);
    float* v_raw  = (float*)alloc((size_t)CT * (CNKV * CD) * 4);
    bf16*  q_bf   = (bf16*)alloc((size_t)CT * (CNH * CD) * 2);
    bf16*  k_bf   = (bf16*)alloc((size_t)CT * (CNKV * CD) * 2);
    bf16*  v_bf   = (bf16*)alloc((size_t)CT * (CNKV * CD) * 2);
    bf16*  o_bf   = (bf16*)alloc((size_t)CT * (CNH * CD) * 2);
    float* oproj  = (float*)alloc((size_t)CT * CH * 4);
    bf16*  h2_bf  = (bf16*)alloc((size_t)CT * CH * 2);
    int*   tids   = (int*)alloc((size_t)CT * CK * 4);
    float* tw     = (float*)alloc((size_t)CT * CK * 4);
    int*   es     = (int*)alloc((size_t)CP * 4);
    int*   pc     = (int*)alloc((size_t)CP * 4);
    bf16*  buf    = (bf16*)alloc((size_t)CE * CC * CH * 2);
    float* gu     = (float*)alloc((size_t)CE * CC * (2 * CF) * 4);
    bf16*  act    = (bf16*)alloc((size_t)CE * CC * CF * 2);
    float* dn     = (float*)alloc((size_t)CE * CC * CH * 4);

    // 1) residual add + input RMSNorm  (resid -> d_out residual slot)
    add_rmsnorm_kernel<<<CT, 256, 0, stream>>>(hidden, residual, w_in_ln,
                                               res_out, h_bf);
    // 2) weight conversion fp32 -> bf16
    f32_to_bf16_kernel<<<4096, 256, 0, stream>>>(wq, wq_bf, (long long)CH * CNH * CD);
    f32_to_bf16_kernel<<<2048, 256, 0, stream>>>(wk, wk_bf, (long long)CH * CNKV * CD);
    f32_to_bf16_kernel<<<2048, 256, 0, stream>>>(wv, wv_bf, (long long)CH * CNKV * CD);
    f32_to_bf16_kernel<<<4096, 256, 0, stream>>>(wo, wo_bf, (long long)CNH * CD * CH);
    f32_to_bf16_kernel<<<8192, 256, 0, stream>>>(w_gate_up, wgu_bf,
                                                 (long long)CE * CH * 2 * CF);
    f32_to_bf16_kernel<<<8192, 256, 0, stream>>>(w_down, wdn_bf,
                                                 (long long)CE * CF * CH);
    // 3) QKV projections (WMMA GEMMs)
    gemm_bf16_wmma<<<dim3((CNH * CD) / 64, CT / 64, 1), 256, 0, stream>>>(
        h_bf, wq_bf, q_raw, CT, CNH * CD, CH, 0, 0, 0);
    gemm_bf16_wmma<<<dim3((CNKV * CD) / 64, CT / 64, 1), 256, 0, stream>>>(
        h_bf, wk_bf, k_raw, CT, CNKV * CD, CH, 0, 0, 0);
    gemm_bf16_wmma<<<dim3((CNKV * CD) / 64, CT / 64, 1), 256, 0, stream>>>(
        h_bf, wv_bf, v_raw, CT, CNKV * CD, CH, 0, 0, 0);
    // 4) Q/K norm + RoPE, V reorder into [b,h,s,d] bf16
    qk_norm_rope_kernel<<<CT * CNH, 128, 0, stream>>>(q_raw, q_norm_w, positions,
                                                      q_bf, CNH);
    qk_norm_rope_kernel<<<CT * CNKV, 128, 0, stream>>>(k_raw, k_norm_w, positions,
                                                       k_bf, CNKV);
    v_reorder_kernel<<<CT * CNKV, 128, 0, stream>>>(v_raw, v_bf);
    // 5) causal flash attention (WMMA)
    flash_attn_kernel<<<dim3(CS / 128, CB * CNH), 256, 0, stream>>>(q_bf, k_bf,
                                                                    v_bf, o_bf);
    // 6) O projection + residual + post RMSNorm
    gemm_bf16_wmma<<<dim3(CH / 64, CT / 64, 1), 256, 0, stream>>>(
        o_bf, wo_bf, oproj, CT, CH, CNH * CD, 0, 0, 0);
    add_rmsnorm_kernel<<<CT, 256, 0, stream>>>(oproj, res_out, w_post_ln,
                                               res_out, h2_bf);
    // 7) MoE routing
    gate_topk_kernel<<<CT, 256, 0, stream>>>(h2_bf, gate_w, tids, tw);
    route_experts_kernel<<<1, 32, 0, stream>>>(tids, es, pc);
    (void)hipMemsetAsync(buf, 0, (size_t)CE * CC * CH * 2, stream);
    gather_tokens_kernel<<<CP, 256, 0, stream>>>(h2_bf, es, pc, buf);
    // 8) expert GEMMs (batched over E via grid.z)
    gemm_bf16_wmma<<<dim3((2 * CF) / 64, CC / 64, CE), 256, 0, stream>>>(
        buf, wgu_bf, gu, CC, 2 * CF, CH,
        (long long)CC * CH, (long long)CH * 2 * CF, (long long)CC * 2 * CF);
    {
        const long long n = (long long)CE * CC * CF;
        silu_mul_kernel<<<(unsigned)((n + 255) / 256), 256, 0, stream>>>(gu, act);
    }
    gemm_bf16_wmma<<<dim3(CH / 64, CC / 64, CE), 256, 0, stream>>>(
        act, wdn_bf, dn, CC, CH, CF,
        (long long)CC * CF, (long long)CF * CH, (long long)CC * CH);
    // 9) weighted combine -> y
    moe_combine_kernel<<<CT, 256, 0, stream>>>(dn, es, pc, tw, y_out);
    (void)in_sizes; (void)n_in; (void)out_size; (void)ws_size;
}
